// GeometryBasedChannel_6313601925843
// MI455X (gfx1250) — matrix operand order repaired and checked by reading, not compiled, not run on hardware
//
#include <hip/hip_runtime.h>
#include <hip/hip_bf16.h>
#include <math.h>

// ---------------------------------------------------------------------------
// Geometry-based channel: H[r,t] = (1/128) * sum_l alpha_l * e^{i*pi*r*sr_l}
//                                             * e^{-i*pi*t*st_l}
// Complex GEMM [1024 x 8192] x [8192 x 1024], f16 WMMA w/ f32 accumulate.
// CDNA5: async global->LDS (ASYNCcnt) + double-buffered LDS + WMMA.
// Karatsuba 3-mult complex product: P1=Br*Ar, P2=Bi*Ai, P3=(Br+Bi)(Ar+Ai);
// Hre = P1-P2, Him = P3-P1-P2  -> 12 WMMA per K-step/wave instead of 16.
// ---------------------------------------------------------------------------

#define NTX   1024      // tx antennas (N dim, columns t)
#define NRX   1024      // rx antennas (M dim, rows r)
#define LRAYS 8192      // K dim (rays)
#define LSHIFT 13       // log2(LRAYS)

#define TM 128          // M tile per workgroup
#define TN 64           // N tile per workgroup
#define TK 32           // K step (one f16 WMMA K)
#define LDSS 40         // LDS row stride in halfs (80B: 16B-aligned, conflict-free)

// LDS arena rows (per buffer): Br[0..127] Bi[128..255] Bs[256..383]
//                              Ar[384..447] Ai[448..511] As[512..575]
#define NROWS 576
#define BUFH  (NROWS * LDSS)        // halfs per buffer
#define BUFB  (BUFH * 2)            // bytes per buffer

typedef __attribute__((ext_vector_type(16))) _Float16 v16h;
typedef __attribute__((ext_vector_type(4)))  _Float16 v4h;
typedef __attribute__((ext_vector_type(8)))  float    v8f;

#define PI_F 3.14159265358979f

// ---------------------------------------------------------------------------
// Kernel 0: per-ray parameter table  ptab[l] = {pi*sin(th_r), pi*sin(th_t),
//                                              alpha_re, alpha_im}
// ---------------------------------------------------------------------------
__global__ void gbc_params(const float* __restrict__ th_t,
                           const float* __restrict__ th_r,
                           const float* __restrict__ a_re,
                           const float* __restrict__ a_im,
                           float4* __restrict__ ptab) {
    int l = blockIdx.x * 256 + threadIdx.x;
    if (l < LRAYS) {
        ptab[l] = make_float4(PI_F * __sinf(th_r[l]),
                              PI_F * __sinf(th_t[l]),
                              a_re[l], a_im[l]);
    }
}

// ---------------------------------------------------------------------------
// Kernel 1: r-side planes  B[r][l] = alpha_l * e^{i*pi*r*sr_l}
// planes: Bre, Bim, Brs = Bre+Bim (sum computed in f32, then f16).
// Layout [r][l] (K contiguous).
// ---------------------------------------------------------------------------
__global__ void gbc_gen_r(const float4* __restrict__ ptab,
                          _Float16* __restrict__ Bre,
                          _Float16* __restrict__ Bim,
                          _Float16* __restrict__ Brs) {
    size_t idx = (size_t)blockIdx.x * 256 + threadIdx.x;   // over NRX*LRAYS
    int r = (int)(idx >> LSHIFT);
    int l = (int)(idx & (LRAYS - 1));
    float4 p = ptab[l];
    float s, c;
    __sincosf(p.x * (float)r, &s, &c);
    float re = p.z * c - p.w * s;
    float im = p.z * s + p.w * c;
    Bre[idx] = (_Float16)re;
    Bim[idx] = (_Float16)im;
    Brs[idx] = (_Float16)(re + im);
}

// ---------------------------------------------------------------------------
// Kernel 2: t-side planes  A[t][l] = e^{-i*pi*t*st_l}  (conjugate)
// planes: Are = cos, Aim = -sin, Ars = cos - sin.
// ---------------------------------------------------------------------------
__global__ void gbc_gen_t(const float4* __restrict__ ptab,
                          _Float16* __restrict__ Are,
                          _Float16* __restrict__ Aim,
                          _Float16* __restrict__ Ars) {
    size_t idx = (size_t)blockIdx.x * 256 + threadIdx.x;   // over NTX*LRAYS
    int t = (int)(idx >> LSHIFT);
    int l = (int)(idx & (LRAYS - 1));
    float4 p = ptab[l];
    float s, c;
    __sincosf(p.y * (float)t, &s, &c);
    Are[idx] = (_Float16)c;
    Aim[idx] = (_Float16)(-s);
    Ars[idx] = (_Float16)(c - s);
}

// ---------------------------------------------------------------------------
// Async global->LDS 16B/lane (GLOBAL_LOAD_ASYNC_TO_LDS_B128, ASYNCcnt).
// ---------------------------------------------------------------------------
__device__ __forceinline__ void async_cp16(unsigned lds_byte_addr,
                                           const _Float16* gsrc) {
    asm volatile("global_load_async_to_lds_b128 %0, %1, off"
                 :: "v"(lds_byte_addr), "v"(gsrc) : "memory");
}

__device__ __forceinline__ void wait_async0() {
    asm volatile("s_wait_asynccnt 0" ::: "memory");
}

// ---------------------------------------------------------------------------
// WMMA 16-bit A-operand (16x32 MxK): lane = (half)*16 + m;
//   lanes 0-15: VGPR0..3 K 0..7,  VGPR4..7 K 16..23
//   lanes16-31: VGPR0..3 K 8..15, VGPR4..7 K 24..31
// WMMA 16-bit B-operand (32x16 KxN): lane = (half)*16 + n;
//   same K order as A: lanes 0-15 K 0..7,16..23; lanes 16-31 K 8..15,24..31.
// (80B LDS rows keep these 16B-aligned -> compiler emits ds_load_b128.)
// ---------------------------------------------------------------------------
__device__ __forceinline__ v16h load_a_frag(const _Float16* row, int kb) {
    v4h x0 = *(const v4h*)(row + kb);
    v4h x1 = *(const v4h*)(row + kb + 4);
    v4h x2 = *(const v4h*)(row + 16 + kb);
    v4h x3 = *(const v4h*)(row + 16 + kb + 4);
    v16h f;
#pragma unroll
    for (int i = 0; i < 4; ++i) {
        f[i] = x0[i]; f[4 + i] = x1[i]; f[8 + i] = x2[i]; f[12 + i] = x3[i];
    }
    return f;
}

__device__ __forceinline__ v16h load_b_frag(const _Float16* p) {
    v4h x0 = *(const v4h*)(p);
    v4h x1 = *(const v4h*)(p + 4);
    v4h x2 = *(const v4h*)(p + 16);
    v4h x3 = *(const v4h*)(p + 20);
    v16h f;
#pragma unroll
    for (int i = 0; i < 4; ++i) {
        f[i] = x0[i]; f[4 + i] = x1[i]; f[8 + i] = x2[i]; f[12 + i] = x3[i];
    }
    return f;
}

// ---------------------------------------------------------------------------
// GEMM kernel: grid (NTX/TN, NRX/TM), 256 threads (8 wave32).
// Double-buffered LDS; async stage of the NEXT K-step issued right after
// the barrier, overlapping this step's 12 WMMAs per wave.
// ---------------------------------------------------------------------------
__global__ __launch_bounds__(256) void gbc_gemm(
        const _Float16* __restrict__ Bre, const _Float16* __restrict__ Bim,
        const _Float16* __restrict__ Brs,
        const _Float16* __restrict__ Are, const _Float16* __restrict__ Aim,
        const _Float16* __restrict__ Ars,
        float2* __restrict__ out) {
    __shared__ _Float16 smem[2][BUFH];

    const int tid  = threadIdx.x;
    const int lane = tid & 31;
    const int wave = tid >> 5;                 // 0..7
    const int r0   = blockIdx.y * TM;
    const int t0   = blockIdx.x * TN;
    const int m0   = wave * 16;

    // ---- precompute per-thread staging rows (wave-uniform trip counts:
    //      waves 0-1 stage 3 rows/lane, waves 2-7 stage 2 rows/lane) ----
    const _Float16* stageSrc[3];
    unsigned        stageOff[3];
    int nrows = 0;
#pragma unroll
    for (int x = 0; x < 3; ++x) {
        int row = tid + x * 256;
        if (row < NROWS) {
            const _Float16* src;
            if (row < 384) {                    // r-side planes
                int plane = row >> 7, rr = row & 127;
                const _Float16* base = (plane == 0) ? Bre
                                     : (plane == 1) ? Bim : Brs;
                src = base + (size_t)(r0 + rr) * LRAYS;
            } else {                            // t-side planes
                int q = row - 384;
                int plane = q >> 6, tt = q & 63;
                const _Float16* base = (plane == 0) ? Are
                                     : (plane == 1) ? Aim : Ars;
                src = base + (size_t)(t0 + tt) * LRAYS;
            }
            stageSrc[nrows] = src;
            stageOff[nrows] = (unsigned)(row * (LDSS * 2));
            ++nrows;
        }
    }
    const unsigned smemBase = (unsigned)(unsigned long long)(uintptr_t)&smem[0][0];

    v8f accP1[4], accP2[4], accP3[4];
    v8f zero = {};
#pragma unroll
    for (int j = 0; j < 4; ++j) { accP1[j] = zero; accP2[j] = zero; accP3[j] = zero; }

    // frag addressing (constant across K loop)
    const int mr   = m0 + (lane & 15);
    const int kb   = (lane >> 4) * 8;          // A-operand K-half select
    const int kb2  = (lane >> 4) * 8;          // B-operand K-half select
    const int nlan = lane & 15;

    // ---- prologue: async-stage K-step 0 into buffer 0 ----
    for (int x = 0; x < nrows; ++x) {
        unsigned d = smemBase + stageOff[x];
#pragma unroll
        for (int i = 0; i < 4; ++i) async_cp16(d + 16 * i, stageSrc[x] + 8 * i);
    }

    for (int k0 = 0; k0 < LRAYS; k0 += TK) {
        const int cur = (k0 >> 5) & 1;

        wait_async0();           // my buffer[cur] transfers complete
        __syncthreads();         // everyone's complete; prior reads retired

        // ---- issue async stage of NEXT K-step into buffer[cur^1] ----
        if (k0 + TK < LRAYS) {
            const unsigned dbase = smemBase + (unsigned)((cur ^ 1) * BUFB);
            for (int x = 0; x < nrows; ++x) {
                const _Float16* s = stageSrc[x] + (k0 + TK);
                unsigned d = dbase + stageOff[x];
#pragma unroll
                for (int i = 0; i < 4; ++i) async_cp16(d + 16 * i, s + 8 * i);
            }
        }

        // ---- A-operand (r-side) frags for this wave ----
        const _Float16* base = &smem[cur][0];
        v16h br = load_a_frag(base + (0   + mr) * LDSS, kb);
        v16h bi = load_a_frag(base + (128 + mr) * LDSS, kb);
        v16h bs = load_a_frag(base + (256 + mr) * LDSS, kb);

        // ---- 4 N-subtiles, 3 WMMAs each (Karatsuba) ----
#pragma unroll
        for (int j = 0; j < 4; ++j) {
            const int nn = j * 16 + nlan;
            v16h ar = load_b_frag(base + (384 + nn) * LDSS + kb2);
            v16h ai = load_b_frag(base + (448 + nn) * LDSS + kb2);
            v16h as = load_b_frag(base + (512 + nn) * LDSS + kb2);
            accP1[j] = __builtin_amdgcn_wmma_f32_16x16x32_f16(
                false, br, false, ar, (short)0, accP1[j], false, false);
            accP2[j] = __builtin_amdgcn_wmma_f32_16x16x32_f16(
                false, bi, false, ai, (short)0, accP2[j], false, false);
            accP3[j] = __builtin_amdgcn_wmma_f32_16x16x32_f16(
                false, bs, false, as, (short)0, accP3[j], false, false);
        }
    }

    // ---- epilogue: Hre = P1-P2, Him = P3-P1-P2;
    //      C layout: VGPR i -> M=i (lanes 0-15) / M=i+8 (lanes 16-31) ----
    const float sc = 0.0078125f;               // 1/sqrt(2*L) == 1/128
#pragma unroll
    for (int j = 0; j < 4; ++j) {
        const int tcol  = t0 + j * 16 + nlan;
        const int rbase = r0 + m0 + ((lane >> 4) * 8);
#pragma unroll
        for (int i = 0; i < 8; ++i) {
            const float p1 = accP1[j][i], p2 = accP2[j][i], p3 = accP3[j][i];
            out[(size_t)(rbase + i) * NTX + tcol] =
                make_float2((p1 - p2) * sc, (p3 - p1 - p2) * sc);
        }
    }
}

// ---------------------------------------------------------------------------
// Launch. Inputs (setup_inputs order): alpha_re, alpha_im, theta_t, theta_r.
// Workspace layout (needs ~96.1 MB):
//   [Bre][Bim][Brs][Are][Aim][Ars] (16MB each) [ptab 128KB]
// Output: complex64 H[1024][1024] as interleaved float pairs.
// ---------------------------------------------------------------------------
extern "C" void kernel_launch(void* const* d_in, const int* in_sizes, int n_in,
                              void* d_out, int out_size, void* d_ws, size_t ws_size,
                              hipStream_t stream) {
    const float* a_re = (const float*)d_in[0];
    const float* a_im = (const float*)d_in[1];
    const float* th_t = (const float*)d_in[2];
    const float* th_r = (const float*)d_in[3];

    char* ws = (char*)d_ws;
    const size_t plane = (size_t)NRX * LRAYS;          // 8 Mi halfs
    _Float16* Bre = (_Float16*)ws;
    _Float16* Bim = Bre + plane;
    _Float16* Brs = Bim + plane;
    _Float16* Are = Brs + plane;
    _Float16* Aim = Are + plane;
    _Float16* Ars = Aim + plane;
    float4*  ptab = (float4*)(ws + 6 * plane * sizeof(_Float16));

    gbc_params<<<LRAYS / 256, 256, 0, stream>>>(th_t, th_r, a_re, a_im, ptab);

    const int genBlocks = (int)(plane / 256);          // 32768
    gbc_gen_r<<<genBlocks, 256, 0, stream>>>(ptab, Bre, Bim, Brs);
    gbc_gen_t<<<genBlocks, 256, 0, stream>>>(ptab, Are, Aim, Ars);

    dim3 grid(NTX / TN, NRX / TM);                     // (16, 8)
    gbc_gemm<<<grid, 256, 0, stream>>>(Bre, Bim, Brs, Are, Aim, Ars,
                                       (float2*)d_out);
}
